// PackingPolicy_10960756540284
// MI455X (gfx1250) — matrix operand order repaired and checked by reading, not compiled, not run on hardware
//
#include <hip/hip_runtime.h>
#include <hip/hip_bf16.h>
#include <cstdint>

// ---------------------------------------------------------------------------
// Types for gfx1250 WMMA (wave32): v_wmma_f32_16x16x32_bf16
// ---------------------------------------------------------------------------
typedef __attribute__((ext_vector_type(16))) __bf16 v16bf;
typedef __attribute__((ext_vector_type(8)))  float  v8f;

typedef unsigned short u16;

static constexpr int Nn  = 8192;   // graph nodes
static constexpr int Cc  = 4096;   // candidates
static constexpr int HID = 128;
static constexpr int EMB = 128;

__device__ __forceinline__ u16 f32_to_bf16(float f) {
  union { float f; uint32_t u; } x; x.f = f;
  uint32_t r = x.u + 0x7FFFu + ((x.u >> 16) & 1u);   // round-to-nearest-even
  return (u16)(r >> 16);
}
__device__ __forceinline__ float bf16_to_f32(u16 b) {
  union { float f; uint32_t u; } x; x.u = ((uint32_t)b) << 16;
  return x.f;
}

union FragU { v16bf v; uint4 q[2]; };

// A fragment (16x32 bf16, row-major source [M x ldA]):
// lane<16 : row=m, K {0..7} in elems 0..7,  {16..23} in elems 8..15
// lane>=16: row=m, K {8..15} and {24..31}
__device__ __forceinline__ v16bf load_a_frag(const u16* __restrict__ A, int ldA,
                                             int row_base, int k0, int lane) {
  const int m  = row_base + (lane & 15);
  const int kk = k0 + ((lane >> 4) << 3);
  const u16* p = A + (size_t)m * (size_t)ldA + kk;
  FragU f;
  f.q[0] = *(const uint4*)(p);
  f.q[1] = *(const uint4*)(p + 16);
  return f.v;
}

// B fragment (32x16 bf16) from a TRANSPOSED operand BT [Nc x K] row-major:
// lane<16 : col=n, K {k0..k0+7},{k0+16..k0+23}; lane>=16: K {k0+8..k0+15},{k0+24..k0+31}
__device__ __forceinline__ v16bf load_b_frag(const u16* __restrict__ BT, int K,
                                             int col_base, int k0, int lane) {
  const int n  = col_base + (lane & 15);
  const int kk = k0 + ((lane >> 4) << 3);
  const u16* p = BT + (size_t)n * (size_t)K + kk;
  FragU f;
  f.q[0] = *(const uint4*)(p);
  f.q[1] = *(const uint4*)(p + 16);
  return f.v;
}

// ---------------------------------------------------------------------------
// Specialized GEMM for agg = (adj @ h) * inv_deg.
// adj is the 128 MB streamed operand: each wave owns a DISTINCT 16-row strip
// (no redundant loads of adj) and computes all 128 output columns (8 WMMA
// tiles), reusing the A fragment from registers 8x per k-step. The redundant
// loads fall on hT (2 MB, L2/WGP$-resident) where they are nearly free.
// Block = 8 waves -> 128 M x 128 N; grid.x = N/128.
// ---------------------------------------------------------------------------
__global__ __launch_bounds__(256)
void agg_gemm_kernel(const u16* __restrict__ A,      // adj bf16 [N x N]
                     const u16* __restrict__ BT,     // hT bf16 [128 x N]
                     const float* __restrict__ row_scale,
                     u16* __restrict__ Crow,         // agg bf16 [N x 128]
                     int Ndim)
{
  const int lane     = threadIdx.x & 31;
  const int wave     = threadIdx.x >> 5;
  const int row_base = blockIdx.x * 128 + wave * 16;

  const int ma = row_base + (lane & 15);
  const int ka = (lane >> 4) << 3;
  const u16* pa = A + (size_t)ma * (size_t)Ndim + ka;

  const int nb = lane & 15;
  const int kb = (lane >> 4) << 3;

  v8f acc[8] = {};

  for (int k0 = 0; k0 < Ndim; k0 += 32) {
    FragU a;
    a.q[0] = *(const uint4*)(pa + k0);
    a.q[1] = *(const uint4*)(pa + k0 + 16);
    __builtin_prefetch(pa + k0 + 64, 0, 1);          // global_prefetch_b8 on adj stream
#pragma unroll
    for (int nt = 0; nt < 8; ++nt) {
      const u16* pb = BT + (size_t)(nt * 16 + nb) * (size_t)Ndim + (kb + k0);
      FragU b;
      b.q[0] = *(const uint4*)(pb);
      b.q[1] = *(const uint4*)(pb + 16);
      acc[nt] = __builtin_amdgcn_wmma_f32_16x16x32_bf16(
          false, a.v, false, b.v, (short)0, acc[nt], false, false);
    }
  }

  const int half  = lane >> 4;
  const int ncol0 = lane & 15;
#pragma unroll
  for (int nt = 0; nt < 8; ++nt) {
#pragma unroll
    for (int r = 0; r < 8; ++r) {
      const int mm = row_base + r + 8 * half;
      const int nn = nt * 16 + ncol0;
      const float v = acc[nt][r] * row_scale[mm];
      Crow[(size_t)mm * HID + nn] = f32_to_bf16(v);
    }
  }
}

// ---------------------------------------------------------------------------
// Generic bf16 WMMA GEMM:  C[M x Nc] = act( A[M x K] * B[K x Nc] + bias )
// A optionally split at Ksplit into (A0 | A1) (concat) where A1 may be a
// broadcast row vector (same for every m). B given transposed: BT [Nc x K].
// Wave tile: 64(M) x 16(N); block: 8 waves -> 64 x 128; grid (M/64, Nc/128).
// Used for the small, L2-resident GEMMs.
// ---------------------------------------------------------------------------
template<bool RELU, bool SCALE_ROW, bool OUT_T, bool OUT_F32, bool CONCAT, bool BCAST2>
__global__ __launch_bounds__(256)
void gemm_bf16_kernel(const u16* __restrict__ A0, int ld0,
                      const u16* __restrict__ A1, int ld1, int Ksplit,
                      const u16* __restrict__ BT,
                      const float* __restrict__ bias,
                      const float* __restrict__ row_scale,
                      u16* __restrict__ Crow, u16* __restrict__ CT,
                      float* __restrict__ Cf32,
                      int M, int K, int Nc)
{
  const int lane     = threadIdx.x & 31;
  const int wave     = threadIdx.x >> 5;
  const int col_base = blockIdx.y * 128 + wave * 16;
  const int row_blk  = blockIdx.x * 64;
  if (col_base >= Nc) return;

  v8f acc[4] = {};

  for (int k0 = 0; k0 < K; k0 += 32) {
    const v16bf b = load_b_frag(BT, K, col_base, k0, lane);
#pragma unroll
    for (int mt = 0; mt < 4; ++mt) {
      const int rb = row_blk + mt * 16;
      v16bf a;
      if (CONCAT && k0 >= Ksplit) {
        if (BCAST2) a = load_a_frag(A1, 0,   rb, k0 - Ksplit, lane);
        else        a = load_a_frag(A1, ld1, rb, k0 - Ksplit, lane);
      } else {
        a = load_a_frag(A0, ld0, rb, k0, lane);
      }
      acc[mt] = __builtin_amdgcn_wmma_f32_16x16x32_bf16(
          false, a, false, b, (short)0, acc[mt], false, false);
    }
  }

  const int n    = col_base + (lane & 15);
  const float bv = bias ? bias[n] : 0.0f;
  const int half = lane >> 4;
#pragma unroll
  for (int mt = 0; mt < 4; ++mt) {
#pragma unroll
    for (int r = 0; r < 8; ++r) {
      const int m = row_blk + mt * 16 + r + 8 * half;
      float v = acc[mt][r];
      if (SCALE_ROW) v *= row_scale[m];
      v += bv;
      if (RELU) v = fmaxf(v, 0.0f);
      const u16 bh = f32_to_bf16(v);
      Crow[(size_t)m * Nc + n] = bh;
      if (OUT_T)   CT[(size_t)n * M + m] = bh;
      if (OUT_F32) Cf32[(size_t)m * Nc + n] = v;
    }
  }
}

// ---------------------------------------------------------------------------
// Small helper kernels
// ---------------------------------------------------------------------------
__global__ void rmax_kernel(const float* __restrict__ rad, float* __restrict__ out, int n) {
  __shared__ float s[256];
  float m = 0.0f;                                    // rad > 0
  for (int i = threadIdx.x; i < n; i += 256) m = fmaxf(m, rad[i]);
  s[threadIdx.x] = m; __syncthreads();
  for (int st = 128; st > 0; st >>= 1) {
    if (threadIdx.x < st) s[threadIdx.x] = fmaxf(s[threadIdx.x], s[threadIdx.x + st]);
    __syncthreads();
  }
  if (threadIdx.x == 0) out[0] = s[0];
}

// per-row: adj f32 -> bf16 copy + inv_deg[row] = 1/max(rowsum, 1)
__global__ __launch_bounds__(256)
void adj_deg_kernel(const float* __restrict__ adj, u16* __restrict__ adjb,
                    float* __restrict__ inv_deg, int n) {
  const int row = blockIdx.x;
  const float* src = adj + (size_t)row * n;
  u16* dst = adjb + (size_t)row * n;
  float s = 0.0f;
  for (int j = threadIdx.x; j < n; j += 256) {
    const float v = src[j];
    s += v;
    dst[j] = f32_to_bf16(v);
  }
  __shared__ float sh[256];
  sh[threadIdx.x] = s; __syncthreads();
  for (int st = 128; st > 0; st >>= 1) {
    if (threadIdx.x < st) sh[threadIdx.x] += sh[threadIdx.x + st];
    __syncthreads();
  }
  if (threadIdx.x == 0) inv_deg[row] = 1.0f / fmaxf(sh[0], 1.0f);
}

// node feats -> h0 = relu(node @ gW0 + gb0), written row-major + transposed bf16
__global__ void build_h0_kernel(const float* __restrict__ pos, const float* __restrict__ rad,
                                const float* __restrict__ Lp, const float* __restrict__ rmax,
                                const float* __restrict__ gW0, const float* __restrict__ gb0,
                                u16* __restrict__ hrow, u16* __restrict__ hT, int n) {
  const int i = blockIdx.x * 256 + threadIdx.x;
  if (i >= n * HID) return;
  const int node = i >> 7, j = i & 127;
  const float invL = 1.0f / Lp[0];
  const float f0 = pos[node * 3 + 0] * invL;
  const float f1 = pos[node * 3 + 1] * invL;
  const float f2 = pos[node * 3 + 2] * invL;
  const float f3 = rad[node] / rmax[0];
  float v = gb0[j] + f0 * gW0[j] + f1 * gW0[128 + j] + f2 * gW0[256 + j] + f3 * gW0[384 + j];
  v = fmaxf(v, 0.0f);
  const u16 b = f32_to_bf16(v);
  hrow[(size_t)node * HID + j] = b;
  hT[(size_t)j * n + node] = b;
}

__global__ void cvt_bf16_kernel(const float* __restrict__ src, u16* __restrict__ dst, int n) {
  const int i = blockIdx.x * 256 + threadIdx.x;
  if (i < n) dst[i] = f32_to_bf16(src[i]);
}

// dst[n][k] = bf16(src[k][n]) : build transposed bf16 weights
__global__ void transpose_cvt_kernel(const float* __restrict__ src, u16* __restrict__ dst,
                                     int K, int Ncols) {
  const int i = blockIdx.x * 256 + threadIdx.x;
  if (i >= K * Ncols) return;
  const int k = i % K, nn = i / K;
  dst[(size_t)nn * K + k] = f32_to_bf16(src[(size_t)k * Ncols + nn]);
}

// stage 1 of column max/mean over h_f32 [n x 128]; 64 rows per block
__global__ __launch_bounds__(256)
void colred_partial_kernel(const float* __restrict__ h, float* __restrict__ pmax,
                           float* __restrict__ psum) {
  const int b = blockIdx.x;
  const int c = threadIdx.x & 127;
  const int half = threadIdx.x >> 7;
  float mx = 0.0f, sm = 0.0f;                        // h >= 0 after relu
  for (int j = 0; j < 32; ++j) {
    const int r = b * 64 + half + 2 * j;
    const float v = h[(size_t)r * HID + c];
    mx = fmaxf(mx, v); sm += v;
  }
  __shared__ float smx[256], ssm[256];
  smx[threadIdx.x] = mx; ssm[threadIdx.x] = sm; __syncthreads();
  if (threadIdx.x < 128) {
    pmax[b * 128 + c] = fmaxf(smx[threadIdx.x], smx[threadIdx.x + 128]);
    psum[b * 128 + c] = ssm[threadIdx.x] + ssm[threadIdx.x + 128];
  }
}

// stage 2 + graph_emb = [max|mean] @ goW + gob  (single block, 128 threads)
__global__ void graph_emb_kernel(const float* __restrict__ pmax, const float* __restrict__ psum,
                                 const float* __restrict__ goW, const float* __restrict__ gob,
                                 float* __restrict__ ge_f32, u16* __restrict__ ge_b,
                                 int nblk, int n) {
  __shared__ float hg[256];
  const int t = threadIdx.x;                         // 0..127
  float mx = 0.0f, sm = 0.0f;
  for (int b = 0; b < nblk; ++b) { mx = fmaxf(mx, pmax[b * 128 + t]); sm += psum[b * 128 + t]; }
  hg[t] = mx;
  hg[128 + t] = sm / (float)n;
  __syncthreads();
  float acc = gob[t];
  for (int k = 0; k < 256; ++k) acc += hg[k] * goW[k * 128 + t];
  ge_f32[t] = acc;
  ge_b[t]   = f32_to_bf16(acc);
}

// final: score[c] = u2[c] . fW3 + fb3 ; mask==0 -> -inf   (one wave32 / candidate)
__global__ __launch_bounds__(256)
void score_kernel(const u16* __restrict__ u2, const float* __restrict__ w3,
                  const float* __restrict__ b3, const int* __restrict__ mask,
                  float* __restrict__ out, int C) {
  const int wave = threadIdx.x >> 5, lane = threadIdx.x & 31;
  const int c = blockIdx.x * 8 + wave;
  if (c >= C) return;
  const u16* row = u2 + (size_t)c * 128;
  float s = 0.0f;
#pragma unroll
  for (int j = 0; j < 4; ++j) {
    const int t = lane + 32 * j;
    s += bf16_to_f32(row[t]) * w3[t];
  }
  for (int off = 16; off > 0; off >>= 1) s += __shfl_down(s, off, 32);
  if (lane == 0) out[c] = (mask[c] == 0) ? -__builtin_inff() : (s + b3[0]);
}

// ---------------------------------------------------------------------------
// Host-side orchestration
// ---------------------------------------------------------------------------
extern "C" void kernel_launch(void* const* d_in, const int* in_sizes, int n_in,
                              void* d_out, int out_size, void* d_ws, size_t ws_size,
                              hipStream_t stream) {
  const float* cand_feat = (const float*)d_in[0];
  const float* graph_pos = (const float*)d_in[1];
  const float* graph_rad = (const float*)d_in[2];
  const float* Lval      = (const float*)d_in[3];
  const float* adj       = (const float*)d_in[4];
  const int*   mask      = (const int*)  d_in[5];
  const float* cW1 = (const float*)d_in[6];  const float* cb1 = (const float*)d_in[7];
  const float* cW2 = (const float*)d_in[8];  const float* cb2 = (const float*)d_in[9];
  const float* gW0 = (const float*)d_in[10]; const float* gb0 = (const float*)d_in[11];
  const float* gnnW = (const float*)d_in[12]; const float* gnnb = (const float*)d_in[13];
  const float* goW = (const float*)d_in[14]; const float* gob = (const float*)d_in[15];
  const float* fW1 = (const float*)d_in[16]; const float* fb1 = (const float*)d_in[17];
  const float* fW2 = (const float*)d_in[18]; const float* fb2 = (const float*)d_in[19];
  const float* fW3 = (const float*)d_in[20]; const float* fb3 = (const float*)d_in[21];
  float* out = (float*)d_out;

  // ---- workspace carve-out (256 B aligned) ----
  uint8_t* ws = (uint8_t*)d_ws;
  size_t off = 0;
  auto take = [&](size_t bytes) -> void* {
    void* p = (void*)(ws + off);
    off += (bytes + 255) & ~(size_t)255;
    return p;
  };
  u16*   adjb    = (u16*)  take((size_t)Nn * Nn * 2);      // 128 MB, L2-resident
  float* inv_deg = (float*)take((size_t)Nn * 4);
  u16*   hrow0   = (u16*)  take((size_t)Nn * HID * 2);
  u16*   hrow1   = (u16*)  take((size_t)Nn * HID * 2);
  u16*   hT0     = (u16*)  take((size_t)Nn * HID * 2);
  u16*   hT1     = (u16*)  take((size_t)Nn * HID * 2);
  u16*   aggb    = (u16*)  take((size_t)Nn * HID * 2);
  float* hf32    = (float*)take((size_t)Nn * HID * 4);
  u16*   candb   = (u16*)  take((size_t)Cc * 32 * 2);
  u16*   t1      = (u16*)  take((size_t)Cc * 128 * 2);
  u16*   ce      = (u16*)  take((size_t)Cc * 128 * 2);
  u16*   u1      = (u16*)  take((size_t)Cc * 256 * 2);
  u16*   u2      = (u16*)  take((size_t)Cc * 128 * 2);
  u16*   gnnWT   = (u16*)  take((size_t)3 * 128 * 256 * 2);
  u16*   cW1T    = (u16*)  take((size_t)128 * 32 * 2);
  u16*   cW2T    = (u16*)  take((size_t)128 * 128 * 2);
  u16*   fW1T    = (u16*)  take((size_t)256 * 256 * 2);
  u16*   fW2T    = (u16*)  take((size_t)128 * 256 * 2);
  float* pmax    = (float*)take((size_t)128 * 128 * 4);
  float* psum    = (float*)take((size_t)128 * 128 * 4);
  float* rmax    = (float*)take(256);
  float* ge_f32  = (float*)take((size_t)128 * 4);
  u16*   ge_b    = (u16*)  take((size_t)128 * 2);
  (void)ws_size; (void)in_sizes; (void)n_in; (void)out_size;

  // ---- graph preprocessing ----
  rmax_kernel<<<1, 256, 0, stream>>>(graph_rad, rmax, Nn);
  adj_deg_kernel<<<Nn, 256, 0, stream>>>(adj, adjb, inv_deg, Nn);
  build_h0_kernel<<<(Nn * HID) / 256, 256, 0, stream>>>(graph_pos, graph_rad, Lval, rmax,
                                                        gW0, gb0, hrow0, hT0, Nn);

  // ---- weight / input conversions ----
  cvt_bf16_kernel<<<(Cc * 32) / 256, 256, 0, stream>>>(cand_feat, candb, Cc * 32);
  transpose_cvt_kernel<<<(32 * 128) / 256, 256, 0, stream>>>(cW1, cW1T, 32, 128);
  transpose_cvt_kernel<<<(128 * 128) / 256, 256, 0, stream>>>(cW2, cW2T, 128, 128);
  for (int i = 0; i < 3; ++i)
    transpose_cvt_kernel<<<(256 * 128) / 256, 256, 0, stream>>>(
        gnnW + (size_t)i * 256 * 128, gnnWT + (size_t)i * 128 * 256, 256, 128);
  transpose_cvt_kernel<<<(256 * 256) / 256, 256, 0, stream>>>(fW1, fW1T, 256, 256);
  transpose_cvt_kernel<<<(256 * 128) / 256, 256, 0, stream>>>(fW2, fW2T, 256, 128);

  // ---- 3 GNN layers (bf16 WMMA, adj resident in L2 after first pass) ----
  u16* hr[2] = { hrow0, hrow1 };
  u16* ht[2] = { hT0,   hT1   };
  int cur = 0;
  for (int i = 0; i < 3; ++i) {
    const int nxt = cur ^ 1;
    // agg = (adj @ h) * inv_deg : specialized kernel, no redundant adj loads
    agg_gemm_kernel<<<Nn / 128, 256, 0, stream>>>(adjb, ht[cur], inv_deg, aggb, Nn);
    // h' = relu([h|agg] @ gnnW[i] + gnnb[i]) : M=N, K=256 (split at 128)
    gemm_bf16_kernel<true, false, true, true, true, false>
        <<<dim3(Nn / 64, 1), 256, 0, stream>>>(
            hr[cur], HID, aggb, HID, HID, gnnWT + (size_t)i * 128 * 256,
            gnnb + (size_t)i * HID, nullptr,
            hr[nxt], ht[nxt], hf32, Nn, 256, HID);
    cur = nxt;
  }

  // ---- global pooling + graph embedding ----
  colred_partial_kernel<<<128, 256, 0, stream>>>(hf32, pmax, psum);
  graph_emb_kernel<<<1, 128, 0, stream>>>(pmax, psum, goW, gob, ge_f32, ge_b, 128, Nn);

  // ---- candidate encoder ----
  gemm_bf16_kernel<true, false, false, false, false, false>
      <<<dim3(Cc / 64, 1), 256, 0, stream>>>(
          candb, 32, nullptr, 0, 0, cW1T, cb1, nullptr,
          t1, nullptr, nullptr, Cc, 32, 128);
  gemm_bf16_kernel<true, false, false, false, false, false>
      <<<dim3(Cc / 64, 1), 256, 0, stream>>>(
          t1, 128, nullptr, 0, 0, cW2T, cb2, nullptr,
          ce, nullptr, nullptr, Cc, 128, 128);

  // ---- fusion MLP (graph_emb broadcast as second half of A) ----
  gemm_bf16_kernel<true, false, false, false, true, true>
      <<<dim3(Cc / 64, 2), 256, 0, stream>>>(
          ce, 128, ge_b, 0, 128, fW1T, fb1, nullptr,
          u1, nullptr, nullptr, Cc, 256, 256);
  gemm_bf16_kernel<true, false, false, false, false, false>
      <<<dim3(Cc / 64, 1), 256, 0, stream>>>(
          u1, 256, nullptr, 0, 0, fW2T, fb2, nullptr,
          u2, nullptr, nullptr, Cc, 256, 128);

  // ---- final scores + mask ----
  score_kernel<<<Cc / 8, 256, 0, stream>>>(u2, fW3, fb3, mask, out, Cc);
}
